// RNNAttention_79190607004164
// MI455X (gfx1250) — compile-verified, matrix operand order checked by reading, not run on hardware
//
#include <hip/hip_runtime.h>
#include <hip/hip_bf16.h>

// ---------------------------------------------------------------------------
// MI455X / gfx1250 implementation.
//  - kernel 1: flash attention over heads 0..7 using v_wmma_f32_16x16x32_f16,
//              with the 64x32 int32 mask tile per key-block fetched by the
//              Tensor Data Mover into double-buffered LDS (tensor_load_to_lds
//              + s_wait_tensorcnt), pipelined one block ahead of the WMMAs.
//  - kernel 2: lockstep batched GRU over value heads 8..11, every step is a
//              (16x64)@(64x192) WMMA GEMM per wave-tile; weights cached in LDS
// ---------------------------------------------------------------------------

typedef __attribute__((ext_vector_type(16))) _Float16 v16h;
typedef __attribute__((ext_vector_type(8)))  _Float16 v8h;
typedef __attribute__((ext_vector_type(8)))  float    v8f;
typedef __attribute__((ext_vector_type(4)))  float    v4f;
typedef unsigned int u32x4 __attribute__((ext_vector_type(4)));
typedef int          i32x4 __attribute__((ext_vector_type(4)));
typedef int          i32x8 __attribute__((ext_vector_type(8)));

#define B_      4
#define H_      12
#define L_      1024
#define DK_     64
#define GLOBAL_ 8
#define LOCAL_  4
#define W_      9
#define NEG_BIG (-1.0e12f)

__device__ __forceinline__ v8f wmma16(v16h a, v16h b, v8f c) {
  // D = A(16x32 f16) * B(32x16 f16) + C(16x16 f32)
  return __builtin_amdgcn_wmma_f32_16x16x32_f16(
      /*neg_a=*/false, a, /*neg_b=*/false, b,
      /*c_mod=*/(short)0, c, /*reuse_a=*/false, /*reuse_b=*/false);
}

// ---------------------------------------------------------------------------
// Tensor Data Mover: 2D tile load, data_size=8B elements, no pad/multicast.
// D# packing per CDNA5 ISA 8.3/8.4 (group0: count/lds/global/type=2;
// group1: data_size, tensor dims == tile dims (tile fully in-bounds),
// tensor_dim0_stride = row pitch). Groups 2/3 zero (2D tile, tile_dim2=0).
// ---------------------------------------------------------------------------
__device__ __forceinline__ void tdm_load_tile_2d(const void* gaddr,
                                                 unsigned lds_addr,
                                                 unsigned d0_u8,   // row len, 8B units
                                                 unsigned d1_rows, // rows
                                                 unsigned pitch_u8)// row pitch, 8B units
{
  unsigned long long ga = (unsigned long long)(size_t)gaddr;
  u32x4 g0;
  g0[0] = 1u;                                             // count=1, user D#
  g0[1] = lds_addr;                                       // LDS byte address
  g0[2] = (unsigned)(ga & 0xFFFFFFFFu);                   // global_addr[31:0]
  g0[3] = (unsigned)((ga >> 32) & 0x01FFFFFFu) | (2u << 30); // [56:32] | type=2
  i32x8 g1;
  g1[0] = (int)(3u << 16);                                // data_size=3 (8B)
  g1[1] = (int)((d0_u8 & 0xFFFFu) << 16);                 // tensor_dim0[15:0]
  g1[2] = (int)((d0_u8 >> 16) | ((d1_rows & 0xFFFFu) << 16)); // d0 hi | d1 lo
  g1[3] = (int)((d1_rows >> 16) | ((d0_u8 & 0xFFFFu) << 16)); // d1 hi | tile_dim0
  g1[4] = (int)(d1_rows & 0xFFFFu);                       // tile_dim1 | tile_dim2=0
  g1[5] = (int)pitch_u8;                                  // tensor_dim0_stride lo32
  g1[6] = 0;                                              // stride0 hi | stride1 lo
  g1[7] = 0;                                              // stride1 hi
  i32x4 z = {};
#if defined(__clang_major__) && (__clang_major__ >= 23)
  i32x8 z8 = {};
  __builtin_amdgcn_tensor_load_to_lds(g0, g1, z, z, z8, 0);
#else
  __builtin_amdgcn_tensor_load_to_lds(g0, g1, z, z, 0);
#endif
}

// Per-lane A/B operand pattern for 16-bit 16x16x32 WMMA (ISA 7.12.2):
// lane supplies K offsets {k0..k0+7, k0+16..k0+23}, k0 = kk + 8*half.
__device__ __forceinline__ v16h ld16_g_f32(const float* __restrict__ row,
                                           int kk, int half, float scale) {
  const v4f* p = (const v4f*)(row + kk + 8 * half);
  v4f x0 = p[0], x1 = p[1], x2 = p[4], x3 = p[5];
  v16h r;
#pragma unroll
  for (int j = 0; j < 4; ++j) {
    r[j]      = (_Float16)(x0[j] * scale);
    r[4 + j]  = (_Float16)(x1[j] * scale);
    r[8 + j]  = (_Float16)(x2[j] * scale);
    r[12 + j] = (_Float16)(x3[j] * scale);
  }
  return r;
}

// Same pattern from an LDS-resident f16 row (two 16B ds reads).
__device__ __forceinline__ v16h ld16_lds_f16(const _Float16* row, int kk, int half) {
  const v8h* p0 = (const v8h*)(row + kk + 8 * half);
  const v8h* p1 = (const v8h*)(row + kk + 8 * half + 16);
  v8h a = *p0, b = *p1;
  v16h r;
#pragma unroll
  for (int j = 0; j < 8; ++j) { r[j] = a[j]; r[8 + j] = b[j]; }
  return r;
}

// B operand where K runs over rows of a row-major matrix (V block for PV).
__device__ __forceinline__ v16h ld_b_vrows(const float* __restrict__ vb,
                                           int kv0, int col, int half) {
  v16h r;
  int k0 = kv0 + 8 * half;
#pragma unroll
  for (int j = 0; j < 8; ++j) {
    r[j]     = (_Float16)vb[(k0 + j) * DK_ + col];
    r[8 + j] = (_Float16)vb[(k0 + 16 + j) * DK_ + col];
  }
  return r;
}

// ---------------------------------------------------------------------------
// Kernel 1: masked softmax attention, heads 0..7.
// grid.x = 32 heads * 16 (64-row q chunks); block = 128 (4 waves);
// each wave owns a 16-row Q tile and streams K/V in 32-key blocks.
// Mask tiles (64 q rows x 32 keys, int32) are DMA'd by the TDM one block
// ahead into double-buffered LDS.
// ---------------------------------------------------------------------------
__global__ __launch_bounds__(128) void attn_kernel(
    const float* __restrict__ q, const float* __restrict__ k,
    const float* __restrict__ v, const int* __restrict__ mask,
    float* __restrict__ out) {
  __shared__ __attribute__((aligned(16))) _Float16 ldsP[4][16 * 32];
  __shared__ __attribute__((aligned(16))) int s_mask[2][64 * 32];

  const int lane = threadIdx.x & 31, wave = threadIdx.x >> 5;
  const int l15 = lane & 15, half = lane >> 4;
  const int head_id = blockIdx.x >> 4;            // 0..31 = b*8+h
  const int b = head_id >> 3, h = head_id & 7;
  const int q0b = (blockIdx.x & 15) * 64;         // block's 64 q rows
  const int q0 = q0b + wave * 16;                 // wave's 16 q rows

  const float* qbase = q + (((b * H_ + h) * L_) + q0) * DK_;
  const float* kbase = k + ((b * H_ + h) * L_) * DK_;
  const float* vbase = v + ((b * H_ + h) * L_) * DK_;
  const int*   mblk  = mask + (b * L_ + q0b) * L_;   // row pitch L_ ints
  _Float16* myP = ldsP[wave];

  const int m = l15;  // A-operand row within the 16-row tile
  // Q pre-scaled by 1/sqrt(dk)=0.125 (exact power of two, lossless in f16).
  v16h qa0 = ld16_g_f32(qbase + m * DK_, 0, half, 0.125f);
  v16h qa1 = ld16_g_f32(qbase + m * DK_, 32, half, 0.125f);

  v8f o[4] = {};                 // 16x64 output accumulator (C-layout)
  float mrow[8], lrow[8];
#pragma unroll
  for (int r = 0; r < 8; ++r) { mrow[r] = -1.0e30f; lrow[r] = 0.0f; }

  const int NB = L_ / 32;
  // Prologue: TDM-load mask tile for block 0. 32 ints/row = 16 x 8B units,
  // 64 rows, pitch 1024 ints = 512 x 8B units.
  if (wave == 0) {
    tdm_load_tile_2d(mblk, (unsigned)(size_t)(void*)&s_mask[0][0], 16u, 64u, 512u);
    __builtin_amdgcn_s_wait_tensorcnt((short)0);
  }
  __syncthreads();

  for (int kb = 0; kb < NB; ++kb) {
    const int kv0 = kb * 32;
    const int cur = kb & 1;
    // Issue next mask tile DMA; it completes under the WMMA work below.
    if (wave == 0 && (kb + 1) < NB)
      tdm_load_tile_2d(mblk + kv0 + 32,
                       (unsigned)(size_t)(void*)&s_mask[cur ^ 1][0], 16u, 64u, 512u);
    if (kb + 1 < NB) {           // prefetch next K/V block (global_prefetch_b8)
      __builtin_prefetch(kbase + (kv0 + 32 + l15) * DK_, 0, 1);
      __builtin_prefetch(vbase + (kv0 + 32 + l15) * DK_, 0, 1);
    }

    // --- scores: two 16x16 tiles over 32 keys --------------------------------
    v8f s[2];
#pragma unroll
    for (int kt = 0; kt < 2; ++kt) {
      const float* krow = kbase + (kv0 + kt * 16 + l15) * DK_;
      v16h b0 = ld16_g_f32(krow, 0, half, 1.0f);
      v16h b1 = ld16_g_f32(krow, 32, half, 1.0f);
      v8f acc = {};
      acc = wmma16(qa0, b0, acc);
      acc = wmma16(qa1, b1, acc);
      s[kt] = acc;
    }

    // --- mask (from TDM-staged LDS tile) + online softmax --------------------
    const int* mtile = &s_mask[cur][wave * 16 * 32];
#pragma unroll
    for (int r = 0; r < 8; ++r) {
      const int qrow = r + 8 * half;
      const int mk0 = mtile[qrow * 32 + l15];
      const int mk1 = mtile[qrow * 32 + 16 + l15];
      float s0 = (mk0 == 0) ? NEG_BIG : s[0][r];
      float s1 = (mk1 == 0) ? NEG_BIG : s[1][r];

      float t = fmaxf(s0, s1);
#pragma unroll
      for (int off = 1; off < 16; off <<= 1) t = fmaxf(t, __shfl_xor(t, off, 32));
      const float mo = mrow[r];
      const float mn = fmaxf(mo, t);
      const float alpha = __expf(mo - mn);
      const float p0 = __expf(s0 - mn), p1 = __expf(s1 - mn);
      float rs = p0 + p1;
#pragma unroll
      for (int off = 1; off < 16; off <<= 1) rs += __shfl_xor(rs, off, 32);
      lrow[r] = lrow[r] * alpha + rs;
      mrow[r] = mn;
#pragma unroll
      for (int nt = 0; nt < 4; ++nt) o[nt][r] *= alpha;
      // stash P (f16) for C-layout -> A-layout conversion
      myP[qrow * 32 + l15]      = (_Float16)p0;
      myP[qrow * 32 + 16 + l15] = (_Float16)p1;
    }
    __threadfence_block();      // DS ops are in-order per wave; keep compiler honest
    v16h pa = ld16_lds_f16(myP + m * 32, 0, half);

    // --- O += P * V ----------------------------------------------------------
#pragma unroll
    for (int nt = 0; nt < 4; ++nt) {
      v16h vb = ld_b_vrows(vbase, kv0, nt * 16 + l15, half);
      o[nt] = wmma16(pa, vb, o[nt]);
    }

    // Retire the in-flight DMA, then publish the fresh tile to all waves.
    if (wave == 0) __builtin_amdgcn_s_wait_tensorcnt((short)0);
    __syncthreads();
  }

  float* obase = out + (((b * 24 + h) * L_) + q0) * DK_;
#pragma unroll
  for (int nt = 0; nt < 4; ++nt)
#pragma unroll
    for (int r = 0; r < 8; ++r) {
      const int qrow = r + 8 * half;
      obase[qrow * DK_ + nt * 16 + l15] = o[nt][r] / lrow[r];
    }
}

// ---------------------------------------------------------------------------
// Kernel 2: lockstep GRU over sliding windows of local value heads.
// grid = (256 row-blocks, 4 param sets g); block = 128 (4 waves).
// ---------------------------------------------------------------------------
__global__ __launch_bounds__(128) void gru_kernel(
    const float* __restrict__ value, const float* __restrict__ w_ih,
    const float* __restrict__ w_hh, const float* __restrict__ b_ih,
    const float* __restrict__ b_hh, float* __restrict__ out) {
  __shared__ __attribute__((aligned(16))) _Float16 s_wih[192 * 64];
  __shared__ __attribute__((aligned(16))) _Float16 s_whh[192 * 64];
  __shared__ float s_bih[192], s_bhh[192];
  __shared__ __attribute__((aligned(16))) _Float16 s_h[4][16 * 64];

  const int g = blockIdx.y;
  const float* wih_g = w_ih + g * 192 * 64;
  const float* whh_g = w_hh + g * 192 * 64;
  for (int i = threadIdx.x; i < 192 * 64; i += 128) {
    s_wih[i] = (_Float16)wih_g[i];
    s_whh[i] = (_Float16)whh_g[i];
  }
  for (int i = threadIdx.x; i < 192; i += 128) {
    s_bih[i] = b_ih[g * 192 + i];
    s_bhh[i] = b_hh[g * 192 + i];
  }
  __syncthreads();

  const int lane = threadIdx.x & 31, wave = threadIdx.x >> 5;
  const int l15 = lane & 15, half = lane >> 4;
  const int rowtile = blockIdx.x * 4 + wave;   // 0..1023, 16 rows each
  const int seg = rowtile >> 6;                // 0..15 = (b, value-head)
  const int bb = seg >> 2, vh = seg & 3;
  const int l0 = (rowtile & 63) * 16;          // base position within segment
  const float* vseg = value + ((bb * H_ + GLOBAL_ + vh) * L_) * DK_;
  _Float16* hbuf = s_h[wave];
  const int m = l15;

  v8f hc[4] = {};  // hidden state, C-layout, 4 column tiles of 16

  for (int t = 0; t < W_; ++t) {
    // stage h as f16 for A-operand reads (wave-private LDS region)
#pragma unroll
    for (int nt = 0; nt < 4; ++nt)
#pragma unroll
      for (int r = 0; r < 8; ++r)
        hbuf[(r + 8 * half) * 64 + nt * 16 + l15] = (_Float16)hc[nt][r];
    __threadfence_block();
    v16h ha0 = ld16_lds_f16(hbuf + m * 64, 0, half);
    v16h ha1 = ld16_lds_f16(hbuf + m * 64, 32, half);

    // window input row: v_pad[l0 + m + t]  (zeros past the segment end)
    const int lr = l0 + m + t;
    v16h va0 = {}, va1 = {};
    if (lr < L_) {
      const float* vr = vseg + lr * DK_;
      va0 = ld16_g_f32(vr, 0, half, 1.0f);
      va1 = ld16_g_f32(vr, 32, half, 1.0f);
    }

    v8f hnew[4];
#pragma unroll
    for (int nt = 0; nt < 4; ++nt) {
      v8f xr = {}, hr = {}, xz = {}, hz = {}, xn = {}, hn = {};
      {
        const int n = nt * 16 + l15;                       // r-gate cols
        v16h b0 = ld16_lds_f16(s_wih + n * 64, 0, half);
        v16h b1 = ld16_lds_f16(s_wih + n * 64, 32, half);
        xr = wmma16(va0, b0, xr); xr = wmma16(va1, b1, xr);
        b0 = ld16_lds_f16(s_whh + n * 64, 0, half);
        b1 = ld16_lds_f16(s_whh + n * 64, 32, half);
        hr = wmma16(ha0, b0, hr); hr = wmma16(ha1, b1, hr);
      }
      {
        const int n = (nt + 4) * 16 + l15;                 // z-gate cols
        v16h b0 = ld16_lds_f16(s_wih + n * 64, 0, half);
        v16h b1 = ld16_lds_f16(s_wih + n * 64, 32, half);
        xz = wmma16(va0, b0, xz); xz = wmma16(va1, b1, xz);
        b0 = ld16_lds_f16(s_whh + n * 64, 0, half);
        b1 = ld16_lds_f16(s_whh + n * 64, 32, half);
        hz = wmma16(ha0, b0, hz); hz = wmma16(ha1, b1, hz);
      }
      {
        const int n = (nt + 8) * 16 + l15;                 // n-gate cols
        v16h b0 = ld16_lds_f16(s_wih + n * 64, 0, half);
        v16h b1 = ld16_lds_f16(s_wih + n * 64, 32, half);
        xn = wmma16(va0, b0, xn); xn = wmma16(va1, b1, xn);
        b0 = ld16_lds_f16(s_whh + n * 64, 0, half);
        b1 = ld16_lds_f16(s_whh + n * 64, 32, half);
        hn = wmma16(ha0, b0, hn); hn = wmma16(ha1, b1, hn);
      }
      const float bi_r = s_bih[nt * 16 + l15],        bh_r = s_bhh[nt * 16 + l15];
      const float bi_z = s_bih[64 + nt * 16 + l15],   bh_z = s_bhh[64 + nt * 16 + l15];
      const float bi_n = s_bih[128 + nt * 16 + l15],  bh_n = s_bhh[128 + nt * 16 + l15];
      v8f hv;
#pragma unroll
      for (int r = 0; r < 8; ++r) {
        const float rg = 1.0f / (1.0f + __expf(-(xr[r] + bi_r + hr[r] + bh_r)));
        const float zg = 1.0f / (1.0f + __expf(-(xz[r] + bi_z + hz[r] + bh_z)));
        const float ng = tanhf(xn[r] + bi_n + rg * (hn[r] + bh_n));
        hv[r] = (1.0f - zg) * ng + zg * hc[nt][r];
      }
      hnew[nt] = hv;
    }
#pragma unroll
    for (int nt = 0; nt < 4; ++nt) hc[nt] = hnew[nt];
  }

  // out channel = 8 + g*4 + vh  (reference reshape/transpose)
  const int ch = 8 + g * 4 + vh;
  float* ob = out + ((bb * 24 + ch) * L_) * DK_;
#pragma unroll
  for (int nt = 0; nt < 4; ++nt)
#pragma unroll
    for (int r = 0; r < 8; ++r) {
      const int lrow = l0 + r + 8 * half;
      ob[lrow * DK_ + nt * 16 + l15] = hc[nt][r];
    }
}

extern "C" void kernel_launch(void* const* d_in, const int* in_sizes, int n_in,
                              void* d_out, int out_size, void* d_ws, size_t ws_size,
                              hipStream_t stream) {
  const float* q    = (const float*)d_in[0];
  const float* k    = (const float*)d_in[1];
  const float* v    = (const float*)d_in[2];
  const int*   mask = (const int*)d_in[3];
  const float* wih  = (const float*)d_in[4];
  const float* whh  = (const float*)d_in[5];
  const float* bih  = (const float*)d_in[6];
  const float* bhh  = (const float*)d_in[7];
  float* out = (float*)d_out;
  (void)in_sizes; (void)n_in; (void)out_size; (void)d_ws; (void)ws_size;

  // heads 0..7: attention. 32 heads * 16 chunks of 64 q-rows, 4 waves/block.
  attn_kernel<<<dim3(512), dim3(128), 0, stream>>>(q, k, v, mask, out);
  // heads 8..23: lockstep GRU. 256 row-blocks x 4 GRU parameter sets.
  gru_kernel<<<dim3(256, 4), dim3(128), 0, stream>>>(v, wih, whh, bih, bhh, out);
}
